// CM_sampler_54443005444430
// MI455X (gfx1250) — matrix operand order repaired and checked by reading, not compiled, not run on hardware
//
#include <hip/hip_runtime.h>

#define NPTS 10000
#define NPAD 10112   // 79 * 128, padded row count
#define DIM  1024
#define CCLS 10
#define PPC  1000
#define TILE 128
#define KSTEP 32
#define LDSW 40      // padded LDS row stride in bf16 elems (80B: 16B-aligned, bank-conflict-free)
#define MATSZ (TILE * LDSW)       // ushorts per matrix slab
#define BUFSZ (4 * MATSZ)         // ushorts per K-slice buffer (A_hi,A_lo,B_hi,B_lo)
#define SMEM_BYTES (2 * BUFSZ * 2)  // double buffered, bytes

typedef __attribute__((ext_vector_type(16))) __bf16 v16bf;
typedef __attribute__((ext_vector_type(8)))  float  v8f;

union BFU  { __bf16 b; unsigned short u; };
union Frag { uint4 u[2]; v16bf v; };

// ---------------------------------------------------------------------------
// Kernel 1: per-row squared norm + split fp32 -> bf16 hi/lo (compensated)
// ---------------------------------------------------------------------------
__global__ void __launch_bounds__(256)
prep_kernel(const float* __restrict__ feats,
            unsigned short* __restrict__ xhi,
            unsigned short* __restrict__ xlo,
            float* __restrict__ sq) {
  const int r = blockIdx.x;
  const int t = threadIdx.x;
  __shared__ float red[256];

  const int k = t * 4;           // 256 threads * 4 = 1024 = DIM exactly
  float s = 0.0f;
  if (r < NPTS) {
    const float* src = feats + (size_t)r * DIM + k;
    float4 x = *(const float4*)src;
    float xs[4] = {x.x, x.y, x.z, x.w};
    unsigned short hs[4], ls[4];
#pragma unroll
    for (int e = 0; e < 4; ++e) {
      BFU h; h.b = (__bf16)xs[e];
      float hf = (float)h.b;
      BFU l; l.b = (__bf16)(xs[e] - hf);
      hs[e] = h.u; ls[e] = l.u;
      s += xs[e] * xs[e];
    }
    *(ushort4*)(xhi + (size_t)r * DIM + k) = make_ushort4(hs[0], hs[1], hs[2], hs[3]);
    *(ushort4*)(xlo + (size_t)r * DIM + k) = make_ushort4(ls[0], ls[1], ls[2], ls[3]);
  } else {
    *(ushort4*)(xhi + (size_t)r * DIM + k) = make_ushort4(0, 0, 0, 0);
    *(ushort4*)(xlo + (size_t)r * DIM + k) = make_ushort4(0, 0, 0, 0);
  }
  red[t] = s;
  __syncthreads();
  for (int off = 128; off > 0; off >>= 1) {
    if (t < off) red[t] += red[t + off];
    __syncthreads();
  }
  if (t == 0) sq[r] = (r < NPTS) ? red[0] : 1e9f;
}

// ---------------------------------------------------------------------------
// Kernel 2: zero the count accumulators
// ---------------------------------------------------------------------------
__global__ void zero_kernel(int* __restrict__ c, int n) {
  int i = blockIdx.x * blockDim.x + threadIdx.x;
  if (i < n) c[i] = 0;
}

// ---------------------------------------------------------------------------
// Async global->LDS staging of one K-slice (4 mats x 128 rows x 32 bf16).
// Each of the 256 threads issues 8 GLOBAL_LOAD_ASYNC_TO_LDS_B128 lanes-worth.
// LDS dest VGPR = low 32 bits of the generic pointer (ISA 10.2: LDS aperture
// address truncates to the LDS byte offset).
// ---------------------------------------------------------------------------
__device__ __forceinline__ void issue_stage(const unsigned short* __restrict__ xhi,
                                            const unsigned short* __restrict__ xlo,
                                            int rowBase, int colBase, int k0,
                                            unsigned short* bufBase, int t) {
#pragma unroll
  for (int it = 0; it < 8; ++it) {
    const int idx = t + it * 256;        // 0..2047 uint4 chunks
    const int mat = idx >> 9;            // 0:A_hi 1:A_lo 2:B_hi 3:B_lo
    const int rem = idx & 511;
    const int row = rem >> 2;            // 0..127
    const int q   = rem & 3;             // 16B chunk within 64B row slice
    const int grow = ((mat < 2) ? rowBase : colBase) + row;
    const unsigned short* gsrc =
        ((mat & 1) ? xlo : xhi) + (size_t)grow * DIM + k0 + q * 8;
    unsigned short* dst = bufBase + mat * MATSZ + row * LDSW + q * 8;
    const unsigned long long ga = (unsigned long long)gsrc;
    const unsigned ldsoff = (unsigned)(unsigned long long)dst;
    asm volatile("global_load_async_to_lds_b128 %0, %1, off"
                 :: "v"(ldsoff), "v"(ga)
                 : "memory");
  }
}

// ---------------------------------------------------------------------------
// Kernel 3: upper-triangle tiled compensated-bf16 Gram + thresholded
// other-class counting. 256 threads = 8 waves, 128x128 block tile, WMMA bf16x3,
// double-buffered async LDS staging. Off-diagonal blocks also emit column sums
// (transpose contribution), halving total WMMA work.
// ---------------------------------------------------------------------------
__global__ void __launch_bounds__(256)
count_kernel(const unsigned short* __restrict__ xhi,
             const unsigned short* __restrict__ xlo,
             const float* __restrict__ sq,
             int* __restrict__ cnts) {
  extern __shared__ unsigned short smem[];   // 2 * BUFSZ ushorts (80 KB)
  __shared__ int colsum[TILE];

  const int rb = blockIdx.y, cb = blockIdx.x;
  if (cb < rb) return;                       // symmetry: upper triangle only
  const int rowBase = rb * TILE;
  const int colBase = cb * TILE;

  // prune tiles fully inside one class on both axes (all pairs excluded)
  {
    const int rc0 = rowBase / PPC, rc1 = (rowBase + TILE - 1) / PPC;
    const int cc0 = colBase / PPC, cc1 = (colBase + TILE - 1) / PPC;
    if (rc0 == rc1 && cc0 == cc1 && rc0 == cc0) return;
  }

  const int t    = threadIdx.x;
  const int lane = t & 31;
  const int wave = t >> 5;          // 0..7 -> 16-row strip
  const int half = lane >> 4;       // 0/1
  const int ln   = lane & 15;

  if (t < TILE) colsum[t] = 0;      // visible after first loop barrier

  v8f acc[8];
#pragma unroll
  for (int n = 0; n < 8; ++n) {
    v8f z = {0.f, 0.f, 0.f, 0.f, 0.f, 0.f, 0.f, 0.f};
    acc[n] = z;
  }

  unsigned short* buf0 = smem;
  unsigned short* buf1 = smem + BUFSZ;

  issue_stage(xhi, xlo, rowBase, colBase, 0, buf0, t);

  const int NK = DIM / KSTEP;       // 32
  for (int ks = 0; ks < NK; ++ks) {
    unsigned short* cur = (ks & 1) ? buf1 : buf0;

    // own wave's previous-slice async loads complete...
    asm volatile("s_wait_asynccnt 0x0" ::: "memory");
    // ...then all waves' loads landed AND everyone is done reading the buffer
    // we are about to overwrite.
    __syncthreads();

    if (ks + 1 < NK)
      issue_stage(xhi, xlo, rowBase, colBase, (ks + 1) * KSTEP,
                  (ks & 1) ? buf0 : buf1, t);

    // A fragments (16x32 bf16, ISA 7.12.2):
    // lane half h, row m=ln: elems 0..7 <- K=8h..8h+7 ; elems 8..15 <- K=16+8h..+7
    Frag ahi, alo;
    {
      const unsigned short* arow = cur + (wave * 16 + ln) * LDSW;
      ahi.u[0] = *(const uint4*)(arow + 8 * half);
      ahi.u[1] = *(const uint4*)(arow + 16 + 8 * half);
      const unsigned short* lrow = arow + MATSZ;
      alo.u[0] = *(const uint4*)(lrow + 8 * half);
      alo.u[1] = *(const uint4*)(lrow + 16 + 8 * half);
    }

#pragma unroll
    for (int nt = 0; nt < 8; ++nt) {
      // B fragment (32x16): lane half h, col n=ln: elems 0..7 <- K=8h..8h+7 ; elems 8..15 <- K=16+8h..+7
      Frag bhi, blo;
      const unsigned short* brow = cur + 2 * MATSZ + (nt * 16 + ln) * LDSW;
      bhi.u[0] = *(const uint4*)(brow + 8 * half);
      bhi.u[1] = *(const uint4*)(brow + 16 + 8 * half);
      const unsigned short* browl = brow + MATSZ;
      blo.u[0] = *(const uint4*)(browl + 8 * half);
      blo.u[1] = *(const uint4*)(browl + 16 + 8 * half);

      acc[nt] = __builtin_amdgcn_wmma_f32_16x16x32_bf16(
          false, ahi.v, false, bhi.v, (short)0, acc[nt], false, false);
      acc[nt] = __builtin_amdgcn_wmma_f32_16x16x32_bf16(
          false, ahi.v, false, blo.v, (short)0, acc[nt], false, false);
      acc[nt] = __builtin_amdgcn_wmma_f32_16x16x32_bf16(
          false, alo.v, false, bhi.v, (short)0, acc[nt], false, false);
    }
  }

  // Epilogue. C layout (16x16 f32): VGPR v, lanes 0-15 -> M=v, lanes 16-31 -> M=v+8
  const int istrip = rowBase + wave * 16 + half * 8;
  float sqi[8]; int clsi[8];
#pragma unroll
  for (int v = 0; v < 8; ++v) {
    const int i = istrip + v;
    sqi[v]  = sq[i];
    clsi[v] = i / PPC;
  }

  int cnt[8]  = {0, 0, 0, 0, 0, 0, 0, 0};
  int ccnt[8] = {0, 0, 0, 0, 0, 0, 0, 0};   // per-nt column partials
#pragma unroll
  for (int nt = 0; nt < 8; ++nt) {
    const int j = colBase + nt * 16 + ln;
    const float sqj = sq[j];
    const int clsj = j / PPC;
#pragma unroll
    for (int v = 0; v < 8; ++v) {
      const float d2 = sqi[v] + sqj - 2.0f * acc[nt][v];
      const int hit = (d2 < 0.25f && clsi[v] != clsj) ? 1 : 0;
      cnt[v]   += hit;
      ccnt[nt] += hit;
    }
  }

  // Row sums: reduce across the 16 lanes of each half (masks stay within half)
#pragma unroll
  for (int v = 0; v < 8; ++v) {
    int c = cnt[v];
    c += __shfl_xor(c, 1);
    c += __shfl_xor(c, 2);
    c += __shfl_xor(c, 4);
    c += __shfl_xor(c, 8);
    if (ln == 0) {
      const int i = istrip + v;
      if (i < NPTS && c != 0) atomicAdd(&cnts[i], c);
    }
  }

  // Column sums (transpose contribution) for off-diagonal blocks only
  const bool offDiag = (rb != cb);
  if (offDiag) {
#pragma unroll
    for (int nt = 0; nt < 8; ++nt) {
      int c = ccnt[nt];
      c += __shfl_xor(c, 16);       // combine the two row-halves
      if (half == 0 && c != 0) atomicAdd(&colsum[nt * 16 + ln], c);
    }
  }
  __syncthreads();
  if (offDiag && t < TILE) {
    const int j = colBase + t;
    const int c = colsum[t];
    if (j < NPTS && c != 0) atomicAdd(&cnts[j], c);
  }
}

// ---------------------------------------------------------------------------
// Kernel 4: per-class stable top-(budget/C) selection by rank counting.
// rank(p) = #{q : c[q] < c[p] or (c[q]==c[p] and q < p)}  == stable argsort pos
// ---------------------------------------------------------------------------
__global__ void __launch_bounds__(256)
select_kernel(const long long* __restrict__ ids,
              const int* __restrict__ budget,
              const int* __restrict__ cnts,
              float* __restrict__ out) {
  const int cls = blockIdx.x;
  const int t = threadIdx.x;
  __shared__ int c[PPC];
  for (int p = t; p < PPC; p += blockDim.x) c[p] = cnts[cls * PPC + p];
  __syncthreads();

  const int bud = budget[0];
  const int pcb = bud / CCLS;
  for (int p = t; p < PPC; p += blockDim.x) {
    const int cp = c[p];
    int rank = 0;
    for (int q = 0; q < PPC; ++q) {
      const int cq = c[q];
      rank += (cq < cp) || (cq == cp && q < p);
    }
    if (rank < pcb) out[cls * pcb + rank] = (float)ids[cls * PPC + p];
    out[bud + cls * PPC + p] = (float)cp;
  }
}

// ---------------------------------------------------------------------------
extern "C" void kernel_launch(void* const* d_in, const int* in_sizes, int n_in,
                              void* d_out, int out_size, void* d_ws, size_t ws_size,
                              hipStream_t stream) {
  const float*     feats  = (const float*)d_in[0];
  const long long* ids    = (const long long*)d_in[1];
  const int*       budget = (const int*)d_in[2];

  unsigned short* xhi = (unsigned short*)d_ws;
  unsigned short* xlo = xhi + (size_t)NPAD * DIM;
  float*          sq  = (float*)(xlo + (size_t)NPAD * DIM);
  int*            cnt = (int*)(sq + NPAD);
  float*          out = (float*)d_out;

  prep_kernel<<<NPAD, 256, 0, stream>>>(feats, xhi, xlo, sq);
  zero_kernel<<<(NPAD + 255) / 256, 256, 0, stream>>>(cnt, NPAD);

  dim3 grid(NPAD / TILE, NPAD / TILE);
  count_kernel<<<grid, 256, SMEM_BYTES, stream>>>(xhi, xlo, sq, cnt);

  select_kernel<<<CCLS, 256, 0, stream>>>(ids, budget, cnt, out);
}
